// TFTWithGNNWrapper_26757646254596
// MI455X (gfx1250) — compile-verified, matrix operand order checked by reading, not run on hardware
//
#include <hip/hip_runtime.h>
#include <hip/hip_bf16.h>
#include <math.h>

// Problem dims (match reference)
#define BB 1024
#define TT 128
#define FF 9
#define HH 256
#define PP 2048
#define NHEADS 4

typedef __attribute__((ext_vector_type(16))) _Float16 v16h;
typedef __attribute__((ext_vector_type(8)))  _Float16 v8h;
typedef __attribute__((ext_vector_type(8)))  float    v8f;

// ---------------------------------------------------------------------------
// Fragment helpers for V_WMMA_F32_16X16X32_F16.
// 16-bit A 16x32 layout (ISA 7.12.2): lane l holds row m=l&15; hi=l>>4.
//   v16h elements 0..7  : k = 8*hi + 0..7
//   v16h elements 8..15 : k = 16 + 8*hi + 0..7
// B (K x N) is symmetric along K with lane holding column n=l&15, so a
// row-major (N,K) weight matrix (how W for x@W.T is stored) is read with two
// contiguous 16-byte loads per lane per 32-wide K step.
// lane_ptr precomputes the per-lane row base; frag_at is then just base + k0.
// ---------------------------------------------------------------------------
__device__ __forceinline__ const _Float16* lane_ptr(const _Float16* __restrict__ base,
                                                    size_t rowstride, size_t row0,
                                                    int lane) {
  return base + (row0 + (size_t)(lane & 15)) * rowstride + (size_t)((lane >> 4) << 3);
}
__device__ __forceinline__ v16h frag_at(const _Float16* __restrict__ lanep, int k0) {
  v8h lo = *(const v8h*)(lanep + k0);
  v8h hi = *(const v8h*)(lanep + k0 + 16);
  return __builtin_shufflevector(lo, hi, 0,1,2,3,4,5,6,7,8,9,10,11,12,13,14,15);
}

// ---------------------------------------------------------------------------
// Generic WMMA GEMM:  out[M,N] = concat(A0[M,K0], A1[M,K1]) * Bw^T + bias
//   Bw is row-major (N, K0+K1) f16 (i.e. weight as stored for x @ W.T).
//   Wave tile: 16(M) x 64(N); block = 8 waves => 32 x 256.
//   act: 0 none, 1 sigmoid, 2 exact gelu. out32/out16 optional.
// Requires: M%32==0, N%256==0, K0%32==0, K1%32==0.
// K-loop is split into two branch-free segments (A0 then A1).
// ---------------------------------------------------------------------------
__global__ __launch_bounds__(256)
void gemm_wmma(const _Float16* __restrict__ A0, long lda0, int K0,
               const _Float16* __restrict__ A1, long lda1, int K1,
               const _Float16* __restrict__ Bw, long ldb,
               const float* __restrict__ bias,
               float* __restrict__ out32, _Float16* __restrict__ out16,
               long ldo, int act) {
  const int lane = threadIdx.x & 31;
  const int wave = threadIdx.x >> 5;
  const int hi   = lane >> 4;
  const size_t m0    = (size_t)blockIdx.y * 32 + (size_t)(wave & 1) * 16;
  const int    nbase = blockIdx.x * 256 + (wave >> 1) * 64;

  const _Float16* bp[4];
#pragma unroll
  for (int nt = 0; nt < 4; ++nt)
    bp[nt] = lane_ptr(Bw, (size_t)ldb, (size_t)(nbase + nt * 16), lane);

  v8f acc[4] = {};

  // ---- segment 1: A0, k in [0, K0) ----
  {
    const _Float16* ap = lane_ptr(A0, (size_t)lda0, m0, lane);
    for (int k0 = 0; k0 < K0; k0 += 32) {
      v16h a  = frag_at(ap, k0);
      v16h b0 = frag_at(bp[0], k0);
      v16h b1 = frag_at(bp[1], k0);
      v16h b2 = frag_at(bp[2], k0);
      v16h b3 = frag_at(bp[3], k0);
      __builtin_prefetch(ap + k0 + 32, 0, 3);
      __builtin_prefetch(bp[0] + k0 + 32, 0, 3);
      __builtin_prefetch(bp[2] + k0 + 32, 0, 3);
      acc[0] = __builtin_amdgcn_wmma_f32_16x16x32_f16(false, a, false, b0, (short)0, acc[0], false, false);
      acc[1] = __builtin_amdgcn_wmma_f32_16x16x32_f16(false, a, false, b1, (short)0, acc[1], false, false);
      acc[2] = __builtin_amdgcn_wmma_f32_16x16x32_f16(false, a, false, b2, (short)0, acc[2], false, false);
      acc[3] = __builtin_amdgcn_wmma_f32_16x16x32_f16(false, a, false, b3, (short)0, acc[3], false, false);
    }
  }
  // ---- segment 2: A1, k in [K0, K0+K1) ----
  if (K1 > 0) {
    const _Float16* ap = lane_ptr(A1, (size_t)lda1, m0, lane);
    for (int k1 = 0; k1 < K1; k1 += 32) {
      v16h a  = frag_at(ap, k1);
      v16h b0 = frag_at(bp[0], K0 + k1);
      v16h b1 = frag_at(bp[1], K0 + k1);
      v16h b2 = frag_at(bp[2], K0 + k1);
      v16h b3 = frag_at(bp[3], K0 + k1);
      acc[0] = __builtin_amdgcn_wmma_f32_16x16x32_f16(false, a, false, b0, (short)0, acc[0], false, false);
      acc[1] = __builtin_amdgcn_wmma_f32_16x16x32_f16(false, a, false, b1, (short)0, acc[1], false, false);
      acc[2] = __builtin_amdgcn_wmma_f32_16x16x32_f16(false, a, false, b2, (short)0, acc[2], false, false);
      acc[3] = __builtin_amdgcn_wmma_f32_16x16x32_f16(false, a, false, b3, (short)0, acc[3], false, false);
    }
  }

  // C/D layout: VGPR r, lanes 0-15 -> row r, lanes 16-31 -> row r+8; col = lane&15
#pragma unroll
  for (int nt = 0; nt < 4; ++nt) {
    const int ncol = nbase + nt * 16 + (lane & 15);
    const float bv = bias ? bias[ncol] : 0.0f;
#pragma unroll
    for (int r = 0; r < 8; ++r) {
      const size_t mrow = m0 + (size_t)r + (size_t)(hi << 3);
      float v = acc[nt][r] + bv;
      if (act == 1)      v = 1.0f / (1.0f + expf(-v));
      else if (act == 2) v = 0.5f * v * (1.0f + erff(v * 0.70710678118f));
      if (out32) out32[mrow * (size_t)ldo + ncol] = v;
      if (out16) out16[mrow * (size_t)ldo + ncol] = (_Float16)v;
    }
  }
}

// ---------------------------------------------------------------------------
// VSN: per (b,t): softmax over 9 selector logits, weighted sum of embeddings.
// ---------------------------------------------------------------------------
__global__ __launch_bounds__(256)
void vsn_kernel(const float* __restrict__ x, const float* __restrict__ sel_w,
                const float* __restrict__ sel_b, const float* __restrict__ emb_w,
                const float* __restrict__ emb_b, _Float16* __restrict__ hout) {
  __shared__ float xs[FF], wts[FF];
  const size_t bt = blockIdx.x;
  const int tid = threadIdx.x;
  if (tid < FF) xs[tid] = x[bt * FF + tid];
  __syncthreads();
  if (tid < FF) {
    float l = sel_b[tid];
    for (int i = 0; i < FF; ++i) l += xs[i] * sel_w[tid * FF + i];
    wts[tid] = l;
  }
  __syncthreads();
  if (tid == 0) {
    float m = wts[0];
    for (int i = 1; i < FF; ++i) m = fmaxf(m, wts[i]);
    float s = 0.f;
    for (int i = 0; i < FF; ++i) { wts[i] = expf(wts[i] - m); s += wts[i]; }
    const float inv = 1.f / s;
    for (int i = 0; i < FF; ++i) wts[i] *= inv;
  }
  __syncthreads();
  float acc = 0.f;
  for (int f = 0; f < FF; ++f)
    acc += wts[f] * (xs[f] * emb_w[f * HH + tid] + emb_b[f * HH + tid]);
  hout[bt * HH + tid] = (_Float16)acc;
}

// ---------------------------------------------------------------------------
// LSTM gate nonlinearity + cell/hidden update (g = xW + hW + b precomputed).
// gate order per reference split: [i | f | g | o]
// ---------------------------------------------------------------------------
__global__ __launch_bounds__(256)
void lstm_update(const float* __restrict__ g, float* __restrict__ c,
                 _Float16* __restrict__ hprev, _Float16* __restrict__ hout,
                 int t) {
  const int b = blockIdx.x, h = threadIdx.x;
  const float* gr = g + (size_t)b * 1024;
  const float gi = gr[h], gf = gr[256 + h], gg = gr[512 + h], go = gr[768 + h];
  float cc = c[(size_t)b * HH + h];
  const float si = 1.f / (1.f + expf(-gi));
  const float sf = 1.f / (1.f + expf(-gf));
  const float so = 1.f / (1.f + expf(-go));
  cc = sf * cc + si * tanhf(gg);
  const float hn = so * tanhf(cc);
  c[(size_t)b * HH + h] = cc;
  const _Float16 h16 = (_Float16)hn;
  hprev[(size_t)b * HH + h] = h16;
  hout[((size_t)b * TT + t) * HH + h] = h16;
}

// ---------------------------------------------------------------------------
// Attention, last-query only. One workgroup per (b, head).
// 8 waves WMMA the 128(T) x 128(K|V) projection (K=256) into LDS,
// then VALU does q(t=127), scores, softmax, weighted V sum.
// ---------------------------------------------------------------------------
__global__ __launch_bounds__(256)
void attn_last(const _Float16* __restrict__ h1, const _Float16* __restrict__ wqkv,
               const float* __restrict__ bqkv, _Float16* __restrict__ o_attn) {
  __shared__ _Float16 kv[128 * 128];  // cols 0..63 = K, 64..127 = V (32 KB)
  __shared__ float qs[64], pe[128], red[2];
  const int b = blockIdx.x >> 2, head = blockIdx.x & 3;
  const int tid = threadIdx.x, lane = tid & 31, wave = tid >> 5, hi = lane >> 4;
  const size_t rowA = (size_t)b * TT + (size_t)wave * 16;

  const _Float16* ap = lane_ptr(h1, HH, rowA, lane);
  const _Float16* bp[8];
#pragma unroll
  for (int nt = 0; nt < 8; ++nt) {
    const int rw = (nt < 4) ? 256 + head * 64 + nt * 16
                            : 512 + head * 64 + (nt - 4) * 16;
    bp[nt] = lane_ptr(wqkv, HH, (size_t)rw, lane);
  }

  v8f acc[8] = {};
  for (int k0 = 0; k0 < HH; k0 += 32) {
    v16h a = frag_at(ap, k0);
#pragma unroll
    for (int nt = 0; nt < 8; ++nt) {
      v16h bf = frag_at(bp[nt], k0);
      acc[nt] = __builtin_amdgcn_wmma_f32_16x16x32_f16(
          false, a, false, bf, (short)0, acc[nt], false, false);
    }
  }
#pragma unroll
  for (int nt = 0; nt < 8; ++nt) {
    const int rwb = (nt < 4) ? 256 + head * 64 + nt * 16
                             : 512 + head * 64 + (nt - 4) * 16;
    const float bv = bqkv[rwb + (lane & 15)];
#pragma unroll
    for (int r = 0; r < 8; ++r) {
      const int t = wave * 16 + r + (hi << 3);
      kv[t * 128 + nt * 16 + (lane & 15)] = (_Float16)(acc[nt][r] + bv);
    }
  }
  __syncthreads();
  if (tid < 64) {  // q at t = T-1 for this head
    const _Float16* hrow = h1 + ((size_t)b * TT + (TT - 1)) * HH;
    const _Float16* wrow = wqkv + (size_t)(head * 64 + tid) * HH;
    float q = bqkv[head * 64 + tid];
    for (int k = 0; k < HH; ++k) q += (float)hrow[k] * (float)wrow[k];
    qs[tid] = q;
  }
  __syncthreads();
  if (tid < 128) {
    float s = 0.f;
    for (int d = 0; d < 64; ++d) s += (float)kv[tid * 128 + d] * qs[d];
    pe[tid] = s * 0.125f;  // 1/sqrt(64)
  }
  __syncthreads();
  if (tid == 0) {
    float m = pe[0];
    for (int i = 1; i < 128; ++i) m = fmaxf(m, pe[i]);
    red[0] = m;
  }
  __syncthreads();
  if (tid < 128) pe[tid] = expf(pe[tid] - red[0]);
  __syncthreads();
  if (tid == 0) {
    float s = 0.f;
    for (int i = 0; i < 128; ++i) s += pe[i];
    red[1] = s;
  }
  __syncthreads();
  if (tid < 64) {
    float o = 0.f;
    for (int t = 0; t < 128; ++t) o += pe[t] * (float)kv[t * 128 + 64 + tid];
    o_attn[(size_t)b * HH + head * 64 + tid] = (_Float16)(o / red[1]);
  }
}

// ---------------------------------------------------------------------------
// LayerNorm over rows of (P, H) + f16 output.
// ---------------------------------------------------------------------------
__global__ __launch_bounds__(256)
void layernorm_k(const float* __restrict__ xin, const float* __restrict__ g,
                 const float* __restrict__ bln, _Float16* __restrict__ yout) {
  __shared__ float red[256];
  const int r = blockIdx.x, tid = threadIdx.x;
  const float v = xin[(size_t)r * HH + tid];
  red[tid] = v;
  __syncthreads();
  for (int s = 128; s > 0; s >>= 1) { if (tid < s) red[tid] += red[tid + s]; __syncthreads(); }
  const float mu = red[0] * (1.0f / HH);
  __syncthreads();
  const float d = v - mu;
  red[tid] = d * d;
  __syncthreads();
  for (int s = 128; s > 0; s >>= 1) { if (tid < s) red[tid] += red[tid + s]; __syncthreads(); }
  const float var = red[0] * (1.0f / HH);
  yout[(size_t)r * HH + tid] = (_Float16)(d * rsqrtf(var + 1e-5f) * g[tid] + bln[tid]);
}

// Final: combined = gate*enr + (1-gate)*hidden ; out[b] = combined . out_w + out_b
__global__ __launch_bounds__(256)
void final_out(const float* __restrict__ gate, const _Float16* __restrict__ enr,
               const _Float16* __restrict__ hid, const float* __restrict__ ow,
               const float* __restrict__ ob, float* __restrict__ out) {
  __shared__ float red[256];
  const int b = blockIdx.x, t = threadIdx.x;
  const float g = gate[(size_t)b * HH + t];
  const float e = (float)enr[(size_t)b * HH + t];
  const float h = (float)hid[(size_t)b * HH + t];
  red[t] = (g * e + (1.f - g) * h) * ow[t];
  __syncthreads();
  for (int s = 128; s > 0; s >>= 1) { if (t < s) red[t] += red[t + s]; __syncthreads(); }
  if (t == 0) out[b] = red[0] + ob[0];
}

// ------------------------------- helpers -----------------------------------
__global__ void cvt_f16(const float* __restrict__ s, _Float16* __restrict__ d, size_t n) {
  size_t i = (size_t)blockIdx.x * blockDim.x + threadIdx.x;
  if (i < n) d[i] = (_Float16)s[i];
}
// Wcomb[n][0:256]=Wih[n], [256:512]=Whh[n]  (4H x 2H, f16)
__global__ void build_wcomb(const float* __restrict__ wih, const float* __restrict__ whh,
                            _Float16* __restrict__ d) {
  size_t i = (size_t)blockIdx.x * blockDim.x + threadIdx.x;
  if (i >= (size_t)1024 * 512) return;
  const size_t n = i >> 9, k = i & 511;
  const float v = (k < 256) ? wih[n * 256 + k] : whh[n * 256 + (k - 256)];
  d[i] = (_Float16)v;
}
__global__ void add_bias2(const float* __restrict__ a, const float* __restrict__ b,
                          float* __restrict__ d, int n) {
  int i = blockIdx.x * blockDim.x + threadIdx.x;
  if (i < n) d[i] = a[i] + b[i];
}
__global__ void degree_k(const float* __restrict__ adj, float* __restrict__ deg) {
  int r = blockIdx.x * blockDim.x + threadIdx.x;
  if (r < PP) {
    float s = 0.f;
    const float* row = adj + (size_t)r * PP;
    for (int c = 0; c < PP; ++c) s += row[c];
    deg[r] = fmaxf(s, 1e-6f);
  }
}
__global__ void adjnorm_k(const float* __restrict__ adj, const float* __restrict__ deg,
                          _Float16* __restrict__ d) {
  size_t i = (size_t)blockIdx.x * blockDim.x + threadIdx.x;
  if (i < (size_t)PP * PP) d[i] = (_Float16)(adj[i] / deg[i >> 11]);
}
// scatter hidden into TRANSPOSED (H x P) embedding table (zeroed beforehand)
__global__ void scatter_k(const int* __restrict__ sku, const _Float16* __restrict__ hid,
                          _Float16* __restrict__ embT) {
  int i = blockIdx.x * blockDim.x + threadIdx.x;
  if (i < BB * HH) {
    const int b = i >> 8, h = i & 255;
    embT[(size_t)h * PP + sku[b]] = hid[(size_t)b * HH + h];
  }
}
__global__ void gather_k(const int* __restrict__ sku, const _Float16* __restrict__ tn,
                         _Float16* __restrict__ enr) {
  int i = blockIdx.x * blockDim.x + threadIdx.x;
  if (i < BB * HH) {
    const int b = i >> 8, h = i & 255;
    enr[(size_t)b * HH + h] = tn[(size_t)sku[b] * HH + h];
  }
}

// ---------------------------------------------------------------------------
extern "C" void kernel_launch(void* const* d_in, const int* in_sizes, int n_in,
                              void* d_out, int out_size, void* d_ws, size_t ws_size,
                              hipStream_t stream) {
  const float* x         = (const float*)d_in[0];
  const int*   sku       = (const int*)d_in[1];
  const float* adj       = (const float*)d_in[2];
  const float* vsn_emb_w = (const float*)d_in[3];
  const float* vsn_emb_b = (const float*)d_in[4];
  const float* vsn_sel_w = (const float*)d_in[5];
  const float* vsn_sel_b = (const float*)d_in[6];
  const float* wih0 = (const float*)d_in[7];
  const float* whh0 = (const float*)d_in[8];
  const float* bih0 = (const float*)d_in[9];
  const float* bhh0 = (const float*)d_in[10];
  const float* wih1 = (const float*)d_in[11];
  const float* whh1 = (const float*)d_in[12];
  const float* bih1 = (const float*)d_in[13];
  const float* bhh1 = (const float*)d_in[14];
  const float* attn_in_w  = (const float*)d_in[15];
  const float* attn_in_b  = (const float*)d_in[16];
  const float* attn_out_w = (const float*)d_in[17];
  const float* attn_out_b = (const float*)d_in[18];
  const float* gnn_w  = (const float*)d_in[19];
  const float* gnn_b  = (const float*)d_in[20];
  const float* ln_g   = (const float*)d_in[21];
  const float* ln_b   = (const float*)d_in[22];
  const float* gate_w = (const float*)d_in[23];
  const float* gate_b = (const float*)d_in[24];
  const float* out_w  = (const float*)d_in[25];
  const float* out_b  = (const float*)d_in[26];
  float* out = (float*)d_out;

  // ---- carve workspace ----
  char* p = (char*)d_ws;
  auto carve = [&](size_t bytes) -> void* {
    void* r = (void*)p;
    p += (bytes + 255) & ~(size_t)255;
    return r;
  };
  const size_t BT = (size_t)BB * TT;
  _Float16* h_vsn   = (_Float16*)carve(BT * HH * 2);
  _Float16* h0      = (_Float16*)carve(BT * HH * 2);
  _Float16* h1      = (_Float16*)carve(BT * HH * 2);
  _Float16* wcomb0  = (_Float16*)carve((size_t)1024 * 512 * 2);
  _Float16* wcomb1  = (_Float16*)carve((size_t)1024 * 512 * 2);
  float*    bcomb0  = (float*)carve(1024 * 4);
  float*    bcomb1  = (float*)carve(1024 * 4);
  _Float16* hprev   = (_Float16*)carve((size_t)BB * HH * 2);
  float*    cst     = (float*)carve((size_t)BB * HH * 4);
  float*    gbuf    = (float*)carve((size_t)BB * 1024 * 4);
  _Float16* wqkv16  = (_Float16*)carve((size_t)768 * HH * 2);
  _Float16* o_attn  = (_Float16*)carve((size_t)BB * HH * 2);
  _Float16* wout16  = (_Float16*)carve((size_t)HH * HH * 2);
  _Float16* hidden16= (_Float16*)carve((size_t)BB * HH * 2);
  _Float16* embT    = (_Float16*)carve((size_t)HH * PP * 2);
  float*    deg     = (float*)carve(PP * 4);
  _Float16* adjn16  = (_Float16*)carve((size_t)PP * PP * 2);
  _Float16* agg16   = (_Float16*)carve((size_t)PP * HH * 2);
  _Float16* wgnn16  = (_Float16*)carve((size_t)HH * HH * 2);
  float*    tbuf    = (float*)carve((size_t)PP * HH * 4);
  _Float16* tnorm16 = (_Float16*)carve((size_t)PP * HH * 2);
  _Float16* enr16   = (_Float16*)carve((size_t)BB * HH * 2);
  _Float16* wgate16 = (_Float16*)carve((size_t)HH * 512 * 2);
  float*    gatebuf = (float*)carve((size_t)BB * HH * 4);
  (void)ws_size; (void)in_sizes; (void)n_in; (void)out_size;

  auto gemm = [&](const _Float16* A0, long lda0, int K0,
                  const _Float16* A1, long lda1, int K1,
                  const _Float16* Bw, long ldb, const float* bias,
                  float* o32, _Float16* o16, long ldo, int M, int N, int act) {
    dim3 grid((unsigned)(N / 256), (unsigned)(M / 32));
    gemm_wmma<<<grid, dim3(256), 0, stream>>>(A0, lda0, K0, A1, lda1, K1,
                                              Bw, ldb, bias, o32, o16, ldo, act);
  };

  // ---- weight prep (f32 -> f16, combined LSTM weights/biases) ----
  build_wcomb<<<(1024 * 512 + 255) / 256, 256, 0, stream>>>(wih0, whh0, wcomb0);
  build_wcomb<<<(1024 * 512 + 255) / 256, 256, 0, stream>>>(wih1, whh1, wcomb1);
  add_bias2<<<4, 256, 0, stream>>>(bih0, bhh0, bcomb0, 1024);
  add_bias2<<<4, 256, 0, stream>>>(bih1, bhh1, bcomb1, 1024);
  cvt_f16<<<(768 * HH + 255) / 256, 256, 0, stream>>>(attn_in_w, wqkv16, (size_t)768 * HH);
  cvt_f16<<<(HH * HH + 255) / 256, 256, 0, stream>>>(attn_out_w, wout16, (size_t)HH * HH);
  cvt_f16<<<(HH * HH + 255) / 256, 256, 0, stream>>>(gnn_w, wgnn16, (size_t)HH * HH);
  cvt_f16<<<(HH * 512 + 255) / 256, 256, 0, stream>>>(gate_w, wgate16, (size_t)HH * 512);

  // ---- VSN ----
  vsn_kernel<<<(unsigned)BT, 256, 0, stream>>>(x, vsn_sel_w, vsn_sel_b,
                                               vsn_emb_w, vsn_emb_b, h_vsn);

  // ---- two LSTM layers: per step, g = [x_t, h_prev] @ Wcomb^T + bcomb ----
  for (int layer = 0; layer < 2; ++layer) {
    const _Float16* hin  = (layer == 0) ? h_vsn : h0;
    _Float16*       hout = (layer == 0) ? h0 : h1;
    const _Float16* wc   = (layer == 0) ? wcomb0 : wcomb1;
    const float*    bc   = (layer == 0) ? bcomb0 : bcomb1;
    hipMemsetAsync(hprev, 0, (size_t)BB * HH * 2, stream);
    hipMemsetAsync(cst, 0, (size_t)BB * HH * 4, stream);
    for (int t = 0; t < TT; ++t) {
      gemm(hin + (size_t)t * HH, (long)TT * HH, HH,   // x_t rows, stride T*H
           hprev, HH, HH,                              // recurrent input
           wc, 2 * HH, bc, gbuf, nullptr, 4 * HH,
           BB, 4 * HH, /*act=*/0);
      lstm_update<<<BB, HH, 0, stream>>>(gbuf, cst, hprev, hout, t);
    }
  }

  // ---- attention (last query only) + output projection ----
  attn_last<<<BB * NHEADS, 256, 0, stream>>>(h1, wqkv16, attn_in_b, o_attn);
  gemm(o_attn, HH, HH, nullptr, 0, 0, wout16, HH, attn_out_b,
       nullptr, hidden16, HH, BB, HH, 0);

  // ---- GNN ----
  hipMemsetAsync(embT, 0, (size_t)HH * PP * 2, stream);
  scatter_k<<<(BB * HH + 255) / 256, 256, 0, stream>>>(sku, hidden16, embT);
  degree_k<<<PP / 256, 256, 0, stream>>>(adj, deg);
  adjnorm_k<<<(unsigned)(((size_t)PP * PP + 255) / 256), 256, 0, stream>>>(adj, deg, adjn16);
  // aggregated = adj_norm @ product_embeds  (B-matrix = transposed embeds, (H,P))
  gemm(adjn16, PP, PP, nullptr, 0, 0, embT, PP, nullptr,
       nullptr, agg16, HH, PP, HH, 0);
  // transformed = gelu(agg @ gnn_w^T + gnn_b)
  gemm(agg16, HH, HH, nullptr, 0, 0, wgnn16, HH, gnn_b,
       tbuf, nullptr, HH, PP, HH, /*gelu*/2);
  layernorm_k<<<PP, 256, 0, stream>>>(tbuf, ln_g, ln_b, tnorm16);
  gather_k<<<(BB * HH + 255) / 256, 256, 0, stream>>>(sku, tnorm16, enr16);

  // ---- gate (dual-A split-K GEMM over [hidden | enriched]) + output ----
  gemm(hidden16, HH, HH, enr16, HH, HH, wgate16, 2 * HH, gate_b,
       gatebuf, nullptr, HH, BB, HH, /*sigmoid*/1);
  final_out<<<BB, 256, 0, stream>>>(gatebuf, enr16, hidden16, out_w, out_b, out);
}
